// HeteroGraphSAGE_57801669869908
// MI455X (gfx1250) — matrix operand order repaired and checked by reading, not compiled, not run on hardware
//
#include <hip/hip_runtime.h>
#include <hip/hip_bf16.h>

// ---------------------------------------------------------------------------
// HeteroGraphSAGE on MI455X (gfx1250, wave32):
//   * all activations + weights pre-converted to bf16 (weights pre-transposed
//     to [N][K] so GEMM staging is pure coalesced b128 -> LDS b128)
//   * single WMMA GEMM kernel: 128x64 block tile, 8 waves, 32x32 per wave,
//     4x v_wmma_f32_16x16x32_bf16 per k-step, optional dual-A pass
//   * edge aggregation: b128 bf16 gathers + fp32 global_atomic_add_f32
//   * layer-2 work restricted to dst < NU (only x[:NU] is returned)
// ---------------------------------------------------------------------------

typedef __attribute__((ext_vector_type(16))) __bf16 v16bf;
typedef __attribute__((ext_vector_type(8)))  float  v8f;

union Frag16 { v16bf v; unsigned u[8]; };

__device__ __forceinline__ float bf2f(unsigned short h) {
  unsigned u = ((unsigned)h) << 16;
  return __builtin_bit_cast(float, u);
}
__device__ __forceinline__ unsigned short f2bf(float f) {
  unsigned u = __builtin_bit_cast(unsigned, f);
  u += 0x7fffu + ((u >> 16) & 1u);          // round-to-nearest-even
  return (unsigned short)(u >> 16);
}

#define TILE_M 128
#define TILE_N 64
#define TILE_K 32
#define LDSS   40   // LDS row stride in ushort: 80B rows -> b128-aligned

__device__ __forceinline__ v8f wmma_bf16(const Frag16& a, const Frag16& b, v8f c) {
  return __builtin_amdgcn_wmma_f32_16x16x32_bf16(
      /*neg_a=*/false, a.v, /*neg_b=*/false, b.v,
      /*c_mod=*/(short)0, c, /*reuse_a=*/false, /*reuse_b=*/false);
}

// A fragment (16x32 bf16): lane half selects K {0..7,16..23} vs {8..15,24..31}
__device__ __forceinline__ void load_a_frag(Frag16& a, const unsigned short* As,
                                            int mrow, int lane) {
  const int half = lane >> 4;
  const unsigned* arow = (const unsigned*)(As + mrow * LDSS);
#pragma unroll
  for (int v = 0; v < 8; ++v) {
    int kb = (v < 4) ? (8 * half + 2 * v) : (16 + 8 * half + 2 * (v - 4));
    a.u[v] = arow[kb >> 1];
  }
}
// B fragment (32x16 bf16): lane half selects K {0..7,16..23} vs {8..15,24..31}
__device__ __forceinline__ void load_b_frag(Frag16& b, const unsigned short* Bs,
                                            int ncol, int lane) {
  const int half = lane >> 4;
  const unsigned* brow = (const unsigned*)(Bs + ncol * LDSS);
#pragma unroll
  for (int v = 0; v < 8; ++v) {
    int kb = (v < 4) ? (8 * half + 2 * v) : (16 + 8 * half + 2 * (v - 4));
    b.u[v] = brow[kb >> 1];
  }
}

// ---- unified bf16 WMMA GEMM: out = act(A0@WT0^T [+ A1@WT1^T] + b) ---------
// A* : [M,K] bf16 row-major ; WT* : [Ncols,K] bf16 row-major (pre-transposed)
__global__ void __launch_bounds__(256)
gemm_bf16_kernel(const unsigned short* __restrict__ A0,
                 const unsigned short* __restrict__ WT0,
                 const unsigned short* __restrict__ A1,
                 const unsigned short* __restrict__ WT1,
                 const float* __restrict__ bias,
                 float* __restrict__ outF, unsigned short* __restrict__ outH,
                 int M, int Ncols, int K, int npass, int doRelu) {
  __shared__ unsigned short As[TILE_M * LDSS];
  __shared__ unsigned short Bs[TILE_N * LDSS];
  const int tid = threadIdx.x, lane = tid & 31, wv = tid >> 5;
  const int wm = wv & 3, wn = wv >> 2;     // 4 waves x 32 rows, 2 waves x 32 cols
  const int l15 = lane & 15, half = lane >> 4;
  const int m0 = blockIdx.y * TILE_M, n0 = blockIdx.x * TILE_N;
  v8f acc00 = {}, acc01 = {}, acc10 = {}, acc11 = {};

  for (int pass = 0; pass < npass; ++pass) {
    const unsigned short* Ag = pass ? A1 : A0;
    const unsigned short* Wg = pass ? WT1 : WT0;
    for (int k0 = 0; k0 < K; k0 += TILE_K) {
      __syncthreads();
      // A tile 128x32 bf16: 512 b128 transfers, 2 per thread, branch-free
#pragma unroll
      for (int i = 0; i < 2; ++i) {
        int e = tid + i * 256;
        int r = e >> 2, c8 = (e & 3) * 8;
        int gr = m0 + r;
        gr = (gr < M) ? gr : (M - 1);      // clamp: garbage rows never stored
        uint4 v4 = *(const uint4*)(Ag + (size_t)gr * K + k0 + c8);
        *(uint4*)(As + r * LDSS + c8) = v4;
      }
      // B tile 64x32 bf16 from WT rows: 256 b128 transfers, 1 per thread
      {
        int n = tid >> 2, c8 = (tid & 3) * 8;
        uint4 v4 = *(const uint4*)(Wg + (size_t)(n0 + n) * K + k0 + c8);
        *(uint4*)(Bs + n * LDSS + c8) = v4;
      }
      __syncthreads();
      Frag16 a0, a1, b0, b1;
      load_a_frag(a0, As, wm * 32 + l15,      lane);
      load_a_frag(a1, As, wm * 32 + 16 + l15, lane);
      load_b_frag(b0, Bs, wn * 32 + l15,      lane);
      load_b_frag(b1, Bs, wn * 32 + 16 + l15, lane);
      acc00 = wmma_bf16(a0, b0, acc00);
      acc01 = wmma_bf16(a0, b1, acc01);
      acc10 = wmma_bf16(a1, b0, acc10);
      acc11 = wmma_bf16(a1, b1, acc11);
    }
  }
#pragma unroll
  for (int mt = 0; mt < 2; ++mt) {
#pragma unroll
    for (int nt = 0; nt < 2; ++nt) {
      const v8f acc = mt ? (nt ? acc11 : acc10) : (nt ? acc01 : acc00);
      int col = n0 + wn * 32 + nt * 16 + l15;
      float bv = bias[col];
#pragma unroll
      for (int vv = 0; vv < 8; ++vv) {
        int row = m0 + wm * 32 + mt * 16 + half * 8 + vv;
        if (row < M) {
          float f = acc[vv] + bv;
          if (doRelu) f = fmaxf(f, 0.f);
          if (outF) outF[(size_t)row * Ncols + col] = f;
          else      outH[(size_t)row * Ncols + col] = f2bf(f);
        }
      }
    }
  }
}

// ---------------- prep kernels ---------------------------------------------
__global__ void f32_to_bf16_kernel(const float* __restrict__ in,
                                   unsigned short* __restrict__ out, size_t nPairs) {
  size_t i = (size_t)blockIdx.x * blockDim.x + threadIdx.x;
  size_t stride = (size_t)gridDim.x * blockDim.x;
  for (size_t p = i; p < nPairs; p += stride) {
    float2 v = ((const float2*)in)[p];
    ((unsigned*)out)[p] = ((unsigned)f2bf(v.y) << 16) | (unsigned)f2bf(v.x);
  }
}

// WT[n*K + k] = bf16(W[k*N + n]);  K is a power of two (64/256/512)
__global__ void transpose_bf16_kernel(const float* __restrict__ W,
                                      unsigned short* __restrict__ WT,
                                      int kLog2, int Ncols, int total) {
  int o = blockIdx.x * blockDim.x + threadIdx.x;
  if (o < total) {
    int k = o & ((1 << kLog2) - 1);
    int n = o >> kLog2;
    WT[o] = f2bf(W[(size_t)k * Ncols + n]);
  }
}

// ---------------- edge kernels ---------------------------------------------
__global__ void zero_f32_kernel(float* __restrict__ p, size_t n4) { // floats/4
  size_t i = (size_t)blockIdx.x * blockDim.x + threadIdx.x;
  size_t stride = (size_t)gridDim.x * blockDim.x;
  float4 z = {0.f, 0.f, 0.f, 0.f};
  for (size_t idx = i; idx < n4; idx += stride) ((float4*)p)[idx] = z;
}

__global__ void degree_kernel(const int* __restrict__ dst, float* __restrict__ cnt,
                              int E) {
  int i = blockIdx.x * blockDim.x + threadIdx.x;
  if (i < E) atomicAdd(&cnt[dst[i]], 1.0f);
}

// one wave per edge: 32 lanes sweep 512 bf16 features as 64 b128 gathers,
// fp32 atomic adds into agg[dst] (L2-side atomics)
__global__ void scatter_kernel(const unsigned short* __restrict__ xh,
                               const int* __restrict__ src,
                               const int* __restrict__ dst,
                               const float* __restrict__ w,
                               float* __restrict__ agg, int E, int dstLimit) {
  int gwave = (int)(((size_t)blockIdx.x * blockDim.x + threadIdx.x) >> 5);
  int lane = threadIdx.x & 31;
  int nwaves = (int)(((size_t)gridDim.x * blockDim.x) >> 5);
  for (int e = gwave; e < E; e += nwaves) {
    int d = dst[e];
    if (d >= dstLimit) continue;           // uniform per wave
    int s = src[e];
    float we = w[e];
    const uint4* xrow = (const uint4*)(xh + (size_t)s * 512);
    float* arow = agg + (size_t)d * 512;
#pragma unroll
    for (int i = 0; i < 2; ++i) {
      int q = lane + i * 32;               // 64 uint4 = 512 bf16
      uint4 p = xrow[q];
      float* a8 = arow + q * 8;
      unsigned d0 = p.x, d1 = p.y, d2 = p.z, d3 = p.w;
      atomicAdd(a8 + 0, we * bf2f((unsigned short)(d0 & 0xffffu)));
      atomicAdd(a8 + 1, we * bf2f((unsigned short)(d0 >> 16)));
      atomicAdd(a8 + 2, we * bf2f((unsigned short)(d1 & 0xffffu)));
      atomicAdd(a8 + 3, we * bf2f((unsigned short)(d1 >> 16)));
      atomicAdd(a8 + 4, we * bf2f((unsigned short)(d2 & 0xffffu)));
      atomicAdd(a8 + 5, we * bf2f((unsigned short)(d2 >> 16)));
      atomicAdd(a8 + 6, we * bf2f((unsigned short)(d3 & 0xffffu)));
      atomicAdd(a8 + 7, we * bf2f((unsigned short)(d3 >> 16)));
    }
  }
}

__global__ void normalize_kernel(const float* __restrict__ agg,
                                 const float* __restrict__ cnt,
                                 unsigned short* __restrict__ outH, size_t nPairs) {
  size_t i = (size_t)blockIdx.x * blockDim.x + threadIdx.x;
  size_t stride = (size_t)gridDim.x * blockDim.x;
  for (size_t u = i; u < nPairs; u += stride) {
    size_t r = u >> 8;
    int c = (int)(u & 255) * 2;
    float inv = 1.0f / fmaxf(cnt[r], 1.0f);
    float f0 = agg[r * 512 + c] * inv;
    float f1 = agg[r * 512 + c + 1] * inv;
    ((unsigned*)outH)[u] = ((unsigned)f2bf(f1) << 16) | (unsigned)f2bf(f0);
  }
}

// ---------------------------------------------------------------------------
extern "C" void kernel_launch(void* const* d_in, const int* in_sizes, int n_in,
                              void* d_out, int out_size, void* d_ws, size_t ws_size,
                              hipStream_t stream) {
  (void)in_sizes; (void)n_in; (void)out_size; (void)ws_size;
  const int NU = 100000, NI = 50000, NR = 10000, Nn = 160000;
  const int H = 512, OUTC = 256, EUI = 2000000, EUR = 1000000;

  const float* x_user   = (const float*)d_in[0];
  const float* x_item   = (const float*)d_in[1];
  const float* x_rating = (const float*)d_in[2];
  const int*   ei_ui    = (const int*)d_in[3];   // [2,EUI] flat: src then dst
  const float* ew_ui    = (const float*)d_in[4];
  const int*   ei_ur    = (const int*)d_in[5];
  const float* ew_ur    = (const float*)d_in[6];
  const float* Wu = (const float*)d_in[7],  *bu = (const float*)d_in[8];
  const float* Wi = (const float*)d_in[9],  *bi = (const float*)d_in[10];
  const float* Wr = (const float*)d_in[11], *br = (const float*)d_in[12];
  const float* W1l = (const float*)d_in[13], *W1r = (const float*)d_in[14];
  const float* b1  = (const float*)d_in[15];
  const float* W2l = (const float*)d_in[16], *W2r = (const float*)d_in[17];
  const float* b2  = (const float*)d_in[18];
  float* out = (float*)d_out;

  // ---- scratch layout (byte offsets, all 256B aligned) ----
  char* ws = (char*)d_ws;
  const size_t NHB = (size_t)Nn * H;                              // 81.92M elems
  unsigned short* x_bf16  = (unsigned short*)(ws);                // 2*NHB
  unsigned short* x1_bf16 = (unsigned short*)(ws + 2 * NHB);      // 2*NHB
  float*          aggf    = (float*)        (ws + 4 * NHB);       // 4*NHB
  unsigned short* aggn    = (unsigned short*)(ws + 8 * NHB);      // 2*NHB
  float*          cnt     = (float*)        (ws + 10 * NHB);      // 4*Nn
  size_t off = 10 * NHB + 640000;
  unsigned short* xu_bf16 = (unsigned short*)(ws + off); off += (size_t)NU * 256 * 2;
  unsigned short* xi_bf16 = (unsigned short*)(ws + off); off += (size_t)NI * 256 * 2;
  unsigned short* xr_bf16 = (unsigned short*)(ws + off); off += (size_t)NR * 64 * 2;
  unsigned short* WuT  = (unsigned short*)(ws + off); off += (size_t)512 * 256 * 2;
  unsigned short* WiT  = (unsigned short*)(ws + off); off += (size_t)512 * 256 * 2;
  unsigned short* WrT  = (unsigned short*)(ws + off); off += (size_t)512 * 64 * 2;
  unsigned short* W1lT = (unsigned short*)(ws + off); off += (size_t)512 * 512 * 2;
  unsigned short* W1rT = (unsigned short*)(ws + off); off += (size_t)512 * 512 * 2;
  unsigned short* W2lT = (unsigned short*)(ws + off); off += (size_t)256 * 512 * 2;
  unsigned short* W2rT = (unsigned short*)(ws + off); off += (size_t)256 * 512 * 2;

  const int T = 256;
  // ---- prep: bf16 features + transposed bf16 weights ----
  f32_to_bf16_kernel<<<2048, T, 0, stream>>>(x_user,   xu_bf16, (size_t)NU * 128);
  f32_to_bf16_kernel<<<2048, T, 0, stream>>>(x_item,   xi_bf16, (size_t)NI * 128);
  f32_to_bf16_kernel<<<512,  T, 0, stream>>>(x_rating, xr_bf16, (size_t)NR * 32);
  transpose_bf16_kernel<<<512, T, 0, stream>>>(Wu,  WuT,  8, 512, 512 * 256);
  transpose_bf16_kernel<<<512, T, 0, stream>>>(Wi,  WiT,  8, 512, 512 * 256);
  transpose_bf16_kernel<<<128, T, 0, stream>>>(Wr,  WrT,  6, 512, 512 * 64);
  transpose_bf16_kernel<<<1024, T, 0, stream>>>(W1l, W1lT, 9, 512, 512 * 512);
  transpose_bf16_kernel<<<1024, T, 0, stream>>>(W1r, W1rT, 9, 512, 512 * 512);
  transpose_bf16_kernel<<<512, T, 0, stream>>>(W2l, W2lT, 9, 256, 256 * 512);
  transpose_bf16_kernel<<<512, T, 0, stream>>>(W2r, W2rT, 9, 256, 256 * 512);

  // ---- degree (shared by both layers) ----
  zero_f32_kernel<<<64, T, 0, stream>>>(cnt, (size_t)Nn / 4);
  degree_kernel<<<(EUI + T - 1) / T, T, 0, stream>>>(ei_ui + EUI, cnt, EUI);
  degree_kernel<<<(EUR + T - 1) / T, T, 0, stream>>>(ei_ur + EUR, cnt, EUR);

  // ---- encoders -> x_bf16 [N,512] ----
  gemm_bf16_kernel<<<dim3(H / TILE_N, (NU + TILE_M - 1) / TILE_M), T, 0, stream>>>(
      xu_bf16, WuT, nullptr, nullptr, bu, nullptr, x_bf16, NU, H, 256, 1, 1);
  gemm_bf16_kernel<<<dim3(H / TILE_N, (NI + TILE_M - 1) / TILE_M), T, 0, stream>>>(
      xi_bf16, WiT, nullptr, nullptr, bi, nullptr, x_bf16 + (size_t)NU * H, NI, H, 256, 1, 1);
  gemm_bf16_kernel<<<dim3(H / TILE_N, (NR + TILE_M - 1) / TILE_M), T, 0, stream>>>(
      xr_bf16, WrT, nullptr, nullptr, br, nullptr, x_bf16 + (size_t)(NU + NI) * H, NR, H, 64, 1, 1);

  // ---- layer 1: scatter-mean + dual GEMM + relu -> x1_bf16 ----
  zero_f32_kernel<<<4096, T, 0, stream>>>(aggf, NHB / 4);
  scatter_kernel<<<4096, T, 0, stream>>>(x_bf16, ei_ui, ei_ui + EUI, ew_ui, aggf, EUI, Nn);
  scatter_kernel<<<4096, T, 0, stream>>>(x_bf16, ei_ur, ei_ur + EUR, ew_ur, aggf, EUR, Nn);
  normalize_kernel<<<4096, T, 0, stream>>>(aggf, cnt, aggn, (size_t)Nn * 256);
  gemm_bf16_kernel<<<dim3(H / TILE_N, (Nn + TILE_M - 1) / TILE_M), T, 0, stream>>>(
      aggn, W1lT, x_bf16, W1rT, b1, nullptr, x1_bf16, Nn, H, H, 2, 1);

  // ---- layer 2: only dst<NU matters for the returned rows ----
  zero_f32_kernel<<<4096, T, 0, stream>>>(aggf, (size_t)NU * H / 4);
  scatter_kernel<<<4096, T, 0, stream>>>(x1_bf16, ei_ui, ei_ui + EUI, ew_ui, aggf, EUI, NU);
  scatter_kernel<<<4096, T, 0, stream>>>(x1_bf16, ei_ur, ei_ur + EUR, ew_ur, aggf, EUR, NU);
  normalize_kernel<<<4096, T, 0, stream>>>(aggf, cnt, aggn, (size_t)NU * 256);
  gemm_bf16_kernel<<<dim3(OUTC / TILE_N, (NU + TILE_M - 1) / TILE_M), T, 0, stream>>>(
      aggn, W2lT, x1_bf16, W2rT, b2, out, nullptr, NU, OUTC, H, 2, 0);
}
